// GLMAttention_11227044512164
// MI455X (gfx1250) — matrix operand order repaired and checked by reading, not compiled, not run on hardware
//
#include <hip/hip_runtime.h>
#include <hip/hip_bf16.h>
#include <math.h>

typedef __bf16 bf16;
typedef __attribute__((ext_vector_type(16))) __bf16 v16bf;
typedef __attribute__((ext_vector_type(8)))  __bf16 v8bf;
typedef __attribute__((ext_vector_type(8)))  float  v8f;
typedef __attribute__((ext_vector_type(4)))  unsigned u32x4;
typedef __attribute__((ext_vector_type(8)))  int      i32x8;
typedef __attribute__((ext_vector_type(4)))  int      i32x4;

#define T_TOK 2048
#define H_DIM 4096
#define NH 32
#define NKV 2
#define HD 128
#define RD 64
#define QKV_OUT ((NH + 2 * NKV) * HD)   // 4608
#define Q_COLS (NH * HD)                // 4096
#define KV_COLS (NKV * HD)              // 256
#define EPS_F 1e-5f
#define GROUP (NH / NKV)                // 16

static __device__ __forceinline__ bf16 f2b(float f) {
  unsigned u = __builtin_bit_cast(unsigned, f);
  u += 0x7FFFu + ((u >> 16) & 1u);                // round-to-nearest-even
  unsigned short hbits = (unsigned short)(u >> 16);
  return __builtin_bit_cast(bf16, hbits);
}

static __device__ __forceinline__ v8f wmma_bf16(v16bf a, v16bf b, v8f c) {
  // (neg_a, A, neg_b, B, c_mod, C, reuse_a, reuse_b)
  return __builtin_amdgcn_wmma_f32_16x16x32_bf16(false, a, false, b, (short)0, c,
                                                 false, false);
}

// ---------------------------------------------------------------------------
// Tensor Data Mover: DMA a 2D tile (rows x cols bf16) from global to LDS,
// inserting 4 DWORDs of padding after every 16 DWORDs (64B row -> 80B row,
// i.e. LDS row stride of 40 bf16). Descriptor per CDNA5 D# spec.
// ---------------------------------------------------------------------------
static __device__ __forceinline__ void tdm_load_tile_2d(
    const bf16* tile_start, unsigned lds_byte_addr,
    int tile_cols_elems, int tile_rows,
    int tensor_cols_elems, int tensor_rows) {
  unsigned long long ga = (unsigned long long)(size_t)tile_start;

  // --- D# group 0 (128b): count=1 | lds_addr | global_addr[56:0] | type=2 ---
  u32x4 g0;
  g0[0] = 1u;                                        // count=1, user mode
  g0[1] = lds_byte_addr;                             // lds_addr [63:32]
  g0[2] = (unsigned)(ga & 0xFFFFFFFFu);              // global_addr[31:0]
  g0[3] = (unsigned)((ga >> 32) & 0x01FFFFFFu)       // global_addr[56:32]
        | (2u << 30);                                // type=2 ("image")

  // --- D# group 1 (256b) ---
  i32x8 g1;
  g1[0] = (int)((1u << 16)    // data_size = 1 -> 2 bytes
              | (1u << 20)    // pad_enable
              | (3u << 22)    // pad_interval code 3 -> every 16 DWORDs
              | (3u << 25));  // pad_amount  code 3 -> 4 DWORDs
  // tensor_dim0 [79:48], tensor_dim1 [111:80]
  g1[1] = (int)(((unsigned)tensor_cols_elems & 0xFFFFu) << 16);
  g1[2] = (int)((((unsigned)tensor_cols_elems >> 16) & 0xFFFFu)
              | (((unsigned)tensor_rows & 0xFFFFu) << 16));
  // tensor_dim1[31:16] | tile_dim0 [127:112]
  g1[3] = (int)((((unsigned)tensor_rows >> 16) & 0xFFFFu)
              | (((unsigned)tile_cols_elems & 0xFFFFu) << 16));
  // tile_dim1 [143:128], tile_dim2 [159:144] = 0
  g1[4] = (int)((unsigned)tile_rows & 0xFFFFu);
  // tensor_dim0_stride [207:160] (elements)
  g1[5] = tensor_cols_elems;
  g1[6] = 0;                  // stride0 hi | tensor_dim1_stride lo (unused, 2D)
  g1[7] = 0;

  i32x4 z4 = {0, 0, 0, 0};    // groups 2/3 unused (<=2D tensor)
#if defined(__clang_major__) && (__clang_major__ >= 23)
  i32x8 z8 = {0, 0, 0, 0, 0, 0, 0, 0};
  __builtin_amdgcn_tensor_load_to_lds(g0, g1, z4, z4, z8, 0);
#else
  __builtin_amdgcn_tensor_load_to_lds(g0, g1, z4, z4, 0);
#endif
}

// ---------------------------------------------------------------------------
// f32 -> bf16 elementwise conversion
// ---------------------------------------------------------------------------
__global__ void cvt_f32_bf16(const float* __restrict__ in, bf16* __restrict__ out,
                             int n) {
  int i = (blockIdx.x * blockDim.x + threadIdx.x) * 4;
  if (i + 3 < n) {
    float4 v = *(const float4*)(in + i);
    out[i + 0] = f2b(v.x);
    out[i + 1] = f2b(v.y);
    out[i + 2] = f2b(v.z);
    out[i + 3] = f2b(v.w);
  } else {
    for (; i < n; ++i) out[i] = f2b(in[i]);
  }
}

// ---------------------------------------------------------------------------
// C[M,N] = A[M,K] * B[N,K]^T (+bias), bf16 operands, f32 accumulate.
// Block: 256 threads (8 waves), tile 128x128, K-step 32.
// Tile staging: TDM tensor_load_to_lds, double buffered (TENSORcnt in-order:
// wait <=2 means current buffer complete while next buffer's 2 DMAs fly).
// ---------------------------------------------------------------------------
#define BM 128
#define BN 128
#define BK 32
#define LDA_T (BK + 8)   // bf16 stride = 80B rows (TDM pad 4DW / 16DW)

__global__ __launch_bounds__(256) void gemm_bf16_wmma(
    const bf16* __restrict__ A, const bf16* __restrict__ B,
    const float* __restrict__ bias, float* __restrict__ C,
    int M, int N, int K) {
  __shared__ bf16 As[2][BM * LDA_T];
  __shared__ bf16 Bs[2][BN * LDA_T];

  const int tid  = threadIdx.x;
  const int lane = tid & 31;
  const int wid  = tid >> 5;
  const int l16  = lane & 15;
  const int hf   = lane >> 4;        // which 16-lane half
  const int waveM = wid & 1;         // 2 waves over M (64 rows each)
  const int waveN = wid >> 1;        // 4 waves over N (32 cols each)

  const int m0 = blockIdx.y * BM;
  const int n0 = blockIdx.x * BN;

  const bool issuer = (wid == 0);    // wave-uniform: wave 0 drives the TDM
  const unsigned ldsA0 = (unsigned)(size_t)&As[0][0];
  const unsigned ldsA1 = (unsigned)(size_t)&As[1][0];
  const unsigned ldsB0 = (unsigned)(size_t)&Bs[0][0];
  const unsigned ldsB1 = (unsigned)(size_t)&Bs[1][0];

  v8f vzero = {};
  v8f acc[4][2];
#pragma unroll
  for (int i = 0; i < 4; ++i)
#pragma unroll
    for (int j = 0; j < 2; ++j) acc[i][j] = vzero;

  const int nk = K / BK;
  if (issuer) {  // prologue: stage k-slab 0 into buffer 0
    tdm_load_tile_2d(A + (size_t)m0 * K, ldsA0, BK, BM, K, M);
    tdm_load_tile_2d(B + (size_t)n0 * K, ldsB0, BK, BN, K, N);
  }

  for (int kb = 0; kb < nk; ++kb) {
    const int cur = kb & 1;
    const bf16* Asb = &As[cur][0];
    const bf16* Bsb = &Bs[cur][0];
    if (kb + 1 < nk) {
      if (issuer) {  // stage next k-slab into the other buffer
        const size_t koff = (size_t)(kb + 1) * BK;
        tdm_load_tile_2d(A + (size_t)m0 * K + koff, cur ? ldsA0 : ldsA1,
                         BK, BM, K, M);
        tdm_load_tile_2d(B + (size_t)n0 * K + koff, cur ? ldsB0 : ldsB1,
                         BK, BN, K, N);
      }
      __builtin_amdgcn_s_wait_tensorcnt(2);  // current buffer's DMAs done
    } else {
      __builtin_amdgcn_s_wait_tensorcnt(0);
    }
    __syncthreads();  // publish LDS tile to all 8 waves

    // A fragments (16x32, MxK): lane<16 row=M, K in {hf*8..+7, hf*8+16..+7}
    v16bf af[4];
#pragma unroll
    for (int mi = 0; mi < 4; ++mi) {
      int row = waveM * 64 + mi * 16 + l16;
      const bf16* p = &Asb[row * LDA_T + hf * 8];
      v8bf lo = *(const v8bf*)p;
      v8bf hi = *(const v8bf*)(p + 16);
#pragma unroll
      for (int j = 0; j < 8; ++j) { af[mi][j] = lo[j]; af[mi][8 + j] = hi[j]; }
    }
    // B fragments (32x16, KxN): lane = N column, K in {hf*8..+7, hf*8+16..+7}
    v16bf bfr[2];
#pragma unroll
    for (int ni = 0; ni < 2; ++ni) {
      int col = waveN * 32 + ni * 16 + l16;
      const bf16* p = &Bsb[col * LDA_T + hf * 8];
      v8bf lo = *(const v8bf*)p;
      v8bf hi = *(const v8bf*)(p + 16);
#pragma unroll
      for (int j = 0; j < 8; ++j) { bfr[ni][j] = lo[j]; bfr[ni][8 + j] = hi[j]; }
    }
#pragma unroll
    for (int mi = 0; mi < 4; ++mi)
#pragma unroll
      for (int ni = 0; ni < 2; ++ni)
        acc[mi][ni] = wmma_bf16(af[mi], bfr[ni], acc[mi][ni]);
    __syncthreads();  // all waves done reading before buffer is re-targeted
  }

  // Epilogue: C layout VGPR r -> M = r + hf*8, N = l16
#pragma unroll
  for (int mi = 0; mi < 4; ++mi) {
#pragma unroll
    for (int ni = 0; ni < 2; ++ni) {
      int col  = n0 + waveN * 32 + ni * 16 + l16;
      int rowb = m0 + waveM * 64 + mi * 16 + hf * 8;
      float bv = bias ? bias[col] : 0.0f;
#pragma unroll
      for (int r = 0; r < 8; ++r)
        C[(size_t)(rowb + r) * N + col] = acc[mi][ni][r] + bv;
    }
  }
}

// ---------------------------------------------------------------------------
// Per (token, head) RMSNorm + RoPE; one wave per vector of 128.
// Q -> qb [NH][T][HD] bf16, K -> kb [NKV][T][HD] bf16
// ---------------------------------------------------------------------------
__global__ __launch_bounds__(32) void qk_norm_rope(
    const float* __restrict__ qkv, const int* __restrict__ positions,
    const float* __restrict__ q_ln_w, const float* __restrict__ k_ln_w,
    bf16* __restrict__ qb, bf16* __restrict__ kb) {
  const int t    = blockIdx.x;
  const int hy   = blockIdx.y;   // 0..NH+NKV-1
  const int lane = threadIdx.x;
  const bool isQ = hy < NH;
  const int h    = isQ ? hy : hy - NH;
  const float* src = qkv + (size_t)t * QKV_OUT + (isQ ? h * HD : Q_COLS + h * HD);
  const float* lw  = isQ ? q_ln_w : k_ln_w;

  float x[4];
  float ss = 0.f;
#pragma unroll
  for (int j = 0; j < 4; ++j) {
    x[j] = src[lane * 4 + j];
    ss += x[j] * x[j];
  }
#pragma unroll
  for (int m = 16; m >= 1; m >>= 1) ss += __shfl_xor(ss, m, 32);
  float inv = rsqrtf(ss / (float)HD + EPS_F);
#pragma unroll
  for (int j = 0; j < 4; ++j) x[j] = x[j] * inv * lw[lane * 4 + j];

  if (lane < 16) {  // dims 4*lane..4*lane+3 < RD -> two rotate pairs
    float pos = (float)positions[t];
#pragma unroll
    for (int p2 = 0; p2 < 2; ++p2) {
      int j = lane * 2 + p2;  // pair index
      float freq = __powf(10000.0f, -(float)(2 * j) / (float)RD);
      float ang = pos * freq;
      float s, c;
      __sincosf(ang, &s, &c);
      float x1 = x[p2 * 2 + 0], x2 = x[p2 * 2 + 1];
      x[p2 * 2 + 0] = x1 * c - x2 * s;
      x[p2 * 2 + 1] = x2 * c + x1 * s;
    }
  }
  bf16* dst = isQ ? (qb + ((size_t)h * T_TOK + t) * HD)
                  : (kb + ((size_t)h * T_TOK + t) * HD);
#pragma unroll
  for (int j = 0; j < 4; ++j) dst[lane * 4 + j] = f2b(x[j]);
}

// V slice of qkv -> vt [NKV*HD][T] bf16 (transposed so PV B-operand is contiguous)
__global__ void v_transpose(const float* __restrict__ qkv, bf16* __restrict__ vt) {
  int idx = blockIdx.x * blockDim.x + threadIdx.x;
  if (idx >= T_TOK * KV_COLS) return;
  int t = idx / KV_COLS;
  int c = idx % KV_COLS;  // kv*HD + d
  float v = qkv[(size_t)t * QKV_OUT + Q_COLS + KV_COLS + c];
  vt[(size_t)c * T_TOK + t] = f2b(v);
}

// ---------------------------------------------------------------------------
// Causal flash attention: one wave per (head, 16-query tile).
// Scores: QK^T via 4 chained bf16 WMMAs per 16-key tile; online softmax with
// intra-half shuffles; P re-packed C->A layout through LDS; PV via 8 WMMAs.
// ---------------------------------------------------------------------------
__global__ __launch_bounds__(32) void attn_wmma(
    const bf16* __restrict__ qb, const bf16* __restrict__ kb,
    const bf16* __restrict__ vt, bf16* __restrict__ ao) {
  __shared__ bf16 Pl[16 * 40];
  const int qt   = blockIdx.x;
  const int h    = blockIdx.y;
  const int kv   = h / GROUP;
  const int lane = threadIdx.x;
  const int l16  = lane & 15;
  const int hf   = lane >> 4;
  const int q0   = qt * 16;
  const float scale = 0.08838834764831845f;  // HD^-0.5

  // Q fragments: 4 chunks of K=32 along HD
  v16bf qf[4];
  {
    const bf16* qrow = qb + ((size_t)h * T_TOK + (q0 + l16)) * HD;
#pragma unroll
    for (int c = 0; c < 4; ++c) {
      const bf16* p = qrow + c * 32 + hf * 8;
      v8bf lo = *(const v8bf*)p;
      v8bf hi = *(const v8bf*)(p + 16);
#pragma unroll
      for (int j = 0; j < 8; ++j) { qf[c][j] = lo[j]; qf[c][8 + j] = hi[j]; }
    }
  }

  v8f vzero = {};
  v8f O[8];
#pragma unroll
  for (int n = 0; n < 8; ++n) O[n] = vzero;
  float mrow[8], lrow[8];
#pragma unroll
  for (int r = 0; r < 8; ++r) { mrow[r] = -1e30f; lrow[r] = 0.f; }

  const int kend = q0 + 15;
  for (int kb0 = 0; kb0 <= kend; kb0 += 32) {
    // --- scores: two 16-key tiles ---
    v8f s[2];
#pragma unroll
    for (int tk = 0; tk < 2; ++tk) {
      s[tk] = vzero;
      int key = kb0 + tk * 16 + l16;  // < T always (T % 32 == 0)
      const bf16* krow = kb + ((size_t)kv * T_TOK + key) * HD;
#pragma unroll
      for (int c = 0; c < 4; ++c) {
        const bf16* p = krow + c * 32 + hf * 8;
        v8bf lo = *(const v8bf*)p;
        v8bf hi = *(const v8bf*)(p + 16);
        v16bf kf;
#pragma unroll
        for (int j = 0; j < 8; ++j) { kf[j] = lo[j]; kf[8 + j] = hi[j]; }
        s[tk] = wmma_bf16(qf[c], kf, s[tk]);
      }
    }

    // --- causal mask + online softmax (rows live per 16-lane half) ---
    float pvals[2][8];
#pragma unroll
    for (int r = 0; r < 8; ++r) {
      int qrow = q0 + hf * 8 + r;
      float s0 = (kb0 + l16 <= qrow)      ? s[0][r] * scale : -1e30f;
      float s1 = (kb0 + 16 + l16 <= qrow) ? s[1][r] * scale : -1e30f;
      float mx = fmaxf(s0, s1);
#pragma unroll
      for (int m = 1; m < 16; m <<= 1) mx = fmaxf(mx, __shfl_xor(mx, m, 32));
      float mnew  = fmaxf(mrow[r], mx);
      float alpha = __expf(mrow[r] - mnew);
      float p0 = __expf(s0 - mnew);
      float p1 = __expf(s1 - mnew);
      pvals[0][r] = p0;
      pvals[1][r] = p1;
      float sum = p0 + p1;
#pragma unroll
      for (int m = 1; m < 16; m <<= 1) sum += __shfl_xor(sum, m, 32);
      lrow[r] = lrow[r] * alpha + sum;
      mrow[r] = mnew;
#pragma unroll
      for (int n = 0; n < 8; ++n) O[n][r] *= alpha;
    }

    // --- re-pack P (C layout) into A layout through LDS ---
    __syncthreads();
#pragma unroll
    for (int tk = 0; tk < 2; ++tk)
#pragma unroll
      for (int r = 0; r < 8; ++r)
        Pl[(hf * 8 + r) * 40 + tk * 16 + l16] = f2b(pvals[tk][r]);
    __syncthreads();
    v16bf pf;
    {
      const bf16* p = &Pl[l16 * 40 + hf * 8];
      v8bf lo = *(const v8bf*)p;
      v8bf hi = *(const v8bf*)(p + 16);
#pragma unroll
      for (int j = 0; j < 8; ++j) { pf[j] = lo[j]; pf[8 + j] = hi[j]; }
    }

    // --- PV: B operand lane = hd column, contiguous keys thanks to vt ---
#pragma unroll
    for (int n = 0; n < 8; ++n) {
      int col = n * 16 + l16;
      const bf16* p = vt + ((size_t)kv * HD + col) * T_TOK + kb0 + hf * 8;
      v8bf lo = *(const v8bf*)p;
      v8bf hi = *(const v8bf*)(p + 16);
      v16bf vf;
#pragma unroll
      for (int j = 0; j < 8; ++j) { vf[j] = lo[j]; vf[8 + j] = hi[j]; }
      O[n] = wmma_bf16(pf, vf, O[n]);
    }
  }

  // normalize + write attention output bf16 [T][NH*HD]
#pragma unroll
  for (int r = 0; r < 8; ++r) {
    float invl = 1.0f / lrow[r];
#pragma unroll
    for (int n = 0; n < 8; ++n) O[n][r] *= invl;
  }
#pragma unroll
  for (int n = 0; n < 8; ++n) {
    int col  = h * HD + n * 16 + l16;
    int rowb = q0 + hf * 8;
#pragma unroll
    for (int r = 0; r < 8; ++r)
      ao[(size_t)(rowb + r) * Q_COLS + col] = f2b(O[n][r]);
  }
}

// ---------------------------------------------------------------------------
extern "C" void kernel_launch(void* const* d_in, const int* in_sizes, int n_in,
                              void* d_out, int out_size, void* d_ws, size_t ws_size,
                              hipStream_t stream) {
  (void)in_sizes; (void)n_in; (void)out_size; (void)ws_size;
  const int*   positions = (const int*)d_in[0];
  const float* hs        = (const float*)d_in[1];
  const float* w_qkv     = (const float*)d_in[2];
  const float* b_qkv     = (const float*)d_in[3];
  const float* w_dense   = (const float*)d_in[4];
  const float* q_ln_w    = (const float*)d_in[5];
  const float* k_ln_w    = (const float*)d_in[6];

  char* ws = (char*)d_ws;
  size_t off = 0;
  auto alloc = [&](size_t bytes) {
    void* p = ws + off;
    off += (bytes + 255) & ~(size_t)255;
    return p;
  };

  bf16*  hsb  = (bf16*)alloc((size_t)T_TOK * H_DIM * 2);
  bf16*  wqb  = (bf16*)alloc((size_t)QKV_OUT * H_DIM * 2);
  bf16*  wdb  = (bf16*)alloc((size_t)H_DIM * Q_COLS * 2);
  float* qkvf = (float*)alloc((size_t)T_TOK * QKV_OUT * 4);
  bf16*  qbuf = (bf16*)alloc((size_t)NH * T_TOK * HD * 2);
  bf16*  kbuf = (bf16*)alloc((size_t)NKV * T_TOK * HD * 2);
  bf16*  vtb  = (bf16*)alloc((size_t)NKV * HD * T_TOK * 2);
  bf16*  aob  = (bf16*)alloc((size_t)T_TOK * Q_COLS * 2);

  {
    int n = T_TOK * H_DIM;
    cvt_f32_bf16<<<(n / 4 + 255) / 256, 256, 0, stream>>>(hs, hsb, n);
    n = QKV_OUT * H_DIM;
    cvt_f32_bf16<<<(n / 4 + 255) / 256, 256, 0, stream>>>(w_qkv, wqb, n);
    n = H_DIM * Q_COLS;
    cvt_f32_bf16<<<(n / 4 + 255) / 256, 256, 0, stream>>>(w_dense, wdb, n);
  }
  {
    dim3 grid(QKV_OUT / BN, T_TOK / BM);  // 36 x 16
    gemm_bf16_wmma<<<grid, 256, 0, stream>>>(hsb, wqb, b_qkv, qkvf,
                                             T_TOK, QKV_OUT, H_DIM);
  }
  qk_norm_rope<<<dim3(T_TOK, NH + NKV), 32, 0, stream>>>(qkvf, positions,
                                                         q_ln_w, k_ln_w,
                                                         qbuf, kbuf);
  {
    int n = T_TOK * KV_COLS;
    v_transpose<<<(n + 255) / 256, 256, 0, stream>>>(qkvf, vtb);
  }
  attn_wmma<<<dim3(T_TOK / 16, NH), 32, 0, stream>>>(qbuf, kbuf, vtb, aob);
  {
    dim3 grid(H_DIM / BN, T_TOK / BM);    // 32 x 16
    gemm_bf16_wmma<<<grid, 256, 0, stream>>>(aob, wdb, nullptr, (float*)d_out,
                                             T_TOK, H_DIM, Q_COLS);
  }
}
